// Model_28329604284990
// MI455X (gfx1250) — compile-verified, matrix operand order checked by reading, not run on hardware
//
#include <hip/hip_runtime.h>
#include <stdint.h>

// ---------------------------------------------------------------------------
// Types for CDNA5 WMMA (gfx1250, wave32)
// ---------------------------------------------------------------------------
typedef __bf16 bf16_t;
typedef __attribute__((ext_vector_type(16))) __bf16 v16bf;
typedef __attribute__((ext_vector_type(8)))  float  v8f;
typedef unsigned int u32x4 __attribute__((ext_vector_type(4)));
typedef int          i32x4 __attribute__((ext_vector_type(4)));
typedef int          i32x8 __attribute__((ext_vector_type(8)));

#define BN_SCALE 0.99999500003749968f   // 1/sqrt(1+1e-5), eval-mode BN

#if defined(__has_builtin)
#  if __has_builtin(__builtin_amdgcn_tensor_load_to_lds) && \
      __has_builtin(__builtin_amdgcn_s_wait_tensorcnt)
#    define USE_TDM 1
#  endif
#endif
#ifndef USE_TDM
#  define USE_TDM 0
#endif

// ---------------------------------------------------------------------------
// TDM: stage a 64(rows) x 64(cols, bf16) weight tile into LDS.
// D# per CDNA5 ISA:
//   group0: count=1 | lds_addr | global_addr[56:0] | type=2
//   group1: data_size=2B, tensor_dim0=64, tensor_dim1=64, tile_dim0=64,
//           tile_dim1=64, tensor_dim0_stride=Cin (elements)
// This toolchain's builtin takes 6 args (g0,g1,g2,g3,g4,cpol).
// ---------------------------------------------------------------------------
__device__ __forceinline__ void tdm_stage_b(const bf16_t* gsrc, unsigned ldsAddr, int Cin)
{
#if USE_TDM
  unsigned long long ga = (unsigned long long)(uintptr_t)gsrc;
  u32x4 g0;
  g0[0] = 1u;                                                  // count=1, user mode
  g0[1] = ldsAddr;                                             // lds_addr (bytes)
  g0[2] = (unsigned)(ga & 0xFFFFFFFFu);                        // global_addr[31:0]
  g0[3] = (unsigned)((ga >> 32) & 0x01FFFFFFu) | 0x80000000u;  // addr[56:32] | type=2
  i32x8 g1;
  g1[0] = (int)(1u << 16);    // workgroup_mask=0, data_size=1 (2 bytes)
  g1[1] = (int)(64u << 16);   // tensor_dim0 = 64  (bits 79:48 low half)
  g1[2] = (int)(64u << 16);   // tensor_dim1 = 64  (bits 111:80 low half)
  g1[3] = (int)(64u << 16);   // tile_dim0  = 64   (bits 127:112)
  g1[4] = 64;                 // tile_dim1  = 64   (bits 143:128); tile_dim2=0
  g1[5] = Cin;                // tensor_dim0_stride[31:0] (elements)
  g1[6] = 0;
  g1[7] = 0;
  i32x4 gz; gz[0] = 0; gz[1] = 0; gz[2] = 0; gz[3] = 0;
  i32x8 gz8; gz8[0] = 0; gz8[1] = 0; gz8[2] = 0; gz8[3] = 0;
  gz8[4] = 0; gz8[5] = 0; gz8[6] = 0; gz8[7] = 0;
  __builtin_amdgcn_tensor_load_to_lds(g0, g1, gz, gz, gz8, 0);
#else
  (void)gsrc; (void)ldsAddr; (void)Cin;
#endif
}

// ===========================================================================
// WMMA GEMM:  Y = epilogue( X[M,Cin](bf16) * W[CoutPad64,Cin]^T(bf16) )
// mode 0: relu(g*((acc+b)*BN_SCALE)+be)             (ConvBNReLU)
// mode 1: relu(g*((acc+b)*BN_SCALE)+be + Res[m,n])  (res block tail)
// mode 2: acc + b                                    (plain linear)
//
// Block: 256 threads = 8 waves, 4(M) x 2(N); wave computes 32x32 (2x2 frags)
// via v_wmma_f32_16x16x32_bf16. B tile (64 rows x 64 cols) is staged to LDS
// by the Tensor Data Mover (double-buffered, s_wait_tensorcnt pipelined);
// A streams from global with b128 loads + prefetch. Cin % 64 == 0 required.
// ===========================================================================
__global__ __launch_bounds__(256) void gemm_bf16_wmma(
    const bf16_t* __restrict__ X, const bf16_t* __restrict__ W,
    const float* __restrict__ bias, const float* __restrict__ gamma,
    const float* __restrict__ beta, const bf16_t* __restrict__ Res,
    float* __restrict__ Y32, bf16_t* __restrict__ Ybf,
    int M, int Cin, int Cout, int mode)
{
  __shared__ bf16_t bsm[2][64 * 64];    // 2 x 8KB double-buffered B tile

  const int lane = threadIdx.x & 31;
  const int wave = threadIdx.x >> 5;
  const int l16  = lane & 15;
  const int lh   = lane >> 4;           // 0/1 half of the wave
  const int wm   = wave & 3;            // 4 waves along M
  const int wn   = wave >> 2;           // 2 waves along N
  const int m0   = blockIdx.y * 128 + wm * 32;
  const int nB   = blockIdx.x * 64;     // block N base
  const int n0   = nB + wn * 32;

  // Clamped A-row pointers (stores are guarded; loads stay in-bounds)
  const int mA0 = (m0 + l16      < M) ? (m0 + l16)      : (M - 1);
  const int mA1 = (m0 + 16 + l16 < M) ? (m0 + 16 + l16) : (M - 1);
  const bf16_t* xr0 = X + (size_t)mA0 * Cin + lh * 8;
  const bf16_t* xr1 = X + (size_t)mA1 * Cin + lh * 8;

  v8f acc[2][2];
#pragma unroll
  for (int i = 0; i < 2; ++i)
#pragma unroll
    for (int j = 0; j < 2; ++j)
#pragma unroll
      for (int r = 0; r < 8; ++r) acc[i][j][r] = 0.0f;

  const int nch = Cin >> 6;             // 64-col chunks
  int cur = 0;

  // ---- prologue: stage chunk 0 into bsm[0] ----
#if USE_TDM
  if (wave == 0)
    tdm_stage_b(W + (size_t)nB * Cin, (unsigned)(uintptr_t)(void*)&bsm[0][0], Cin);
#else
  {
    const bf16_t* src = W + (size_t)nB * Cin;
#pragma unroll
    for (int j = 0; j < 2; ++j) {
      int idx = threadIdx.x + j * 256;          // 512 x 16B
      int row = idx >> 3, col8 = (idx & 7) * 8;
      *(uint4*)&bsm[0][row * 64 + col8] = *(const uint4*)&src[(size_t)row * Cin + col8];
    }
  }
#endif

  for (int ci = 0; ci < nch; ++ci) {
    // stage next chunk into the other buffer
    if (ci + 1 < nch) {
#if USE_TDM
      if (wave == 0)
        tdm_stage_b(W + (size_t)nB * Cin + ((size_t)(ci + 1) << 6),
                    (unsigned)(uintptr_t)(void*)&bsm[cur ^ 1][0], Cin);
#else
      const bf16_t* src = W + (size_t)nB * Cin + ((size_t)(ci + 1) << 6);
#pragma unroll
      for (int j = 0; j < 2; ++j) {
        int idx = threadIdx.x + j * 256;
        int row = idx >> 3, col8 = (idx & 7) * 8;
        *(uint4*)&bsm[cur ^ 1][row * 64 + col8] =
            *(const uint4*)&src[(size_t)row * Cin + col8];
      }
#endif
    }
#if USE_TDM
    if (wave == 0) {
      if (ci + 1 < nch) __builtin_amdgcn_s_wait_tensorcnt(1);  // chunk ci landed
      else              __builtin_amdgcn_s_wait_tensorcnt(0);
    }
#endif
    __syncthreads();                    // publish chunk ci

    const int ckk = ci << 6;
    const bf16_t* bbase = &bsm[cur][0];
    const int rw0 = wn * 32 + l16;
    __builtin_prefetch(xr0 + ckk + 256, 0, 1);   // global_prefetch_b8
    __builtin_prefetch(xr1 + ckk + 256, 0, 1);

#pragma unroll
    for (int kk = 0; kk < 64; kk += 32) {
      uint4 a0lo = *(const uint4*)(xr0 + ckk + kk);
      uint4 a0hi = *(const uint4*)(xr0 + ckk + kk + 16);
      uint4 a1lo = *(const uint4*)(xr1 + ckk + kk);
      uint4 a1hi = *(const uint4*)(xr1 + ckk + kk + 16);
      uint4 b0lo = *(const uint4*)&bbase[rw0 * 64 + kk + lh * 8];
      uint4 b0hi = *(const uint4*)&bbase[rw0 * 64 + kk + lh * 8 + 16];
      uint4 b1lo = *(const uint4*)&bbase[(rw0 + 16) * 64 + kk + lh * 8];
      uint4 b1hi = *(const uint4*)&bbase[(rw0 + 16) * 64 + kk + lh * 8 + 16];
      v16bf a0, a1, b0, b1;
      __builtin_memcpy(&a0, &a0lo, 16); __builtin_memcpy((char*)&a0 + 16, &a0hi, 16);
      __builtin_memcpy(&a1, &a1lo, 16); __builtin_memcpy((char*)&a1 + 16, &a1hi, 16);
      __builtin_memcpy(&b0, &b0lo, 16); __builtin_memcpy((char*)&b0 + 16, &b0hi, 16);
      __builtin_memcpy(&b1, &b1lo, 16); __builtin_memcpy((char*)&b1 + 16, &b1hi, 16);
      acc[0][0] = __builtin_amdgcn_wmma_f32_16x16x32_bf16(false, a0, false, b0, (short)0, acc[0][0], false, false);
      acc[0][1] = __builtin_amdgcn_wmma_f32_16x16x32_bf16(false, a0, false, b1, (short)0, acc[0][1], false, false);
      acc[1][0] = __builtin_amdgcn_wmma_f32_16x16x32_bf16(false, a1, false, b0, (short)0, acc[1][0], false, false);
      acc[1][1] = __builtin_amdgcn_wmma_f32_16x16x32_bf16(false, a1, false, b1, (short)0, acc[1][1], false, false);
    }
    __syncthreads();                    // all waves done with bsm[cur]
    cur ^= 1;
  }

  // ---- epilogue: C/D layout VGPR r -> (m = base + r + 8*lh, n = nbase + l16) ----
#pragma unroll
  for (int j = 0; j < 2; ++j) {
    const int n = n0 + j * 16 + l16;
    float bb = 0.f, gg = 0.f, be = 0.f;
    if (n < Cout) {
      bb = bias[n];
      if (mode != 2) { gg = gamma[n]; be = beta[n]; }
    }
#pragma unroll
    for (int i = 0; i < 2; ++i) {
#pragma unroll
      for (int r = 0; r < 8; ++r) {
        const int m = m0 + i * 16 + r + lh * 8;
        const bool ok = (m < M) && (n < Cout);
        float v = acc[i][j][r] + bb;
        float y;
        if (mode == 2) {
          y = v;
        } else {
          y = gg * (v * BN_SCALE) + be;
          if (mode == 1 && ok) y += (float)Res[(size_t)m * Cout + n];
          y = fmaxf(y, 0.0f);
        }
        if (ok) {
          if (Ybf) Ybf[(size_t)m * Cout + n] = (bf16_t)y;
          if (Y32) Y32[(size_t)m * Cout + n] = y;
        }
      }
    }
  }
}

// ===========================================================================
// Support kernels
// ===========================================================================

// x [16,3,N] -> xyz [16,N,3]
__global__ void transpose_xyz_kernel(const float* __restrict__ x,
                                     float* __restrict__ xyz, int N)
{
  int i = blockIdx.x * blockDim.x + threadIdx.x;   // b*N + n
  if (i >= 16 * N) return;
  int b = i / N, n = i % N;
  xyz[(size_t)i * 3 + 0] = x[((size_t)b * 3 + 0) * N + n];
  xyz[(size_t)i * 3 + 1] = x[((size_t)b * 3 + 1) * N + n];
  xyz[(size_t)i * 3 + 2] = x[((size_t)b * 3 + 2) * N + n];
}

// embed: feat[b,n,c] = relu(bn(xyz . W[c,:] + b)), C=64, Cin=3 (tiny: VALU)
__global__ void embed_kernel(const float* __restrict__ xyz,
                             const float* __restrict__ Wm,
                             const float* __restrict__ bias,
                             const float* __restrict__ g,
                             const float* __restrict__ be,
                             bf16_t* __restrict__ feat, int N)
{
  int i = blockIdx.x * blockDim.x + threadIdx.x;   // (b*N+n)*64 + c
  if (i >= 16 * N * 64) return;
  int c = i & 63;
  int pn = i >> 6;
  const float* p = xyz + (size_t)pn * 3;
  float v = p[0] * Wm[c * 3 + 0] + p[1] * Wm[c * 3 + 1] + p[2] * Wm[c * 3 + 2] + bias[c];
  v = g[c] * (v * BN_SCALE) + be[c];
  feat[i] = (bf16_t)fmaxf(v, 0.f);
}

// Farthest point sampling: one block per batch (lax.scan semantics:
// idx[0]=0, then successive argmax of running min-distance, ties -> lowest idx)
__global__ __launch_bounds__(256) void fps_kernel(const float* __restrict__ xyz,
                                                  int N, int S, int* __restrict__ out)
{
  int b = blockIdx.x;
  const float* P = xyz + (size_t)b * N * 3;
  __shared__ float d2[1024];
  __shared__ float rv[256];
  __shared__ int   ri[256];
  __shared__ float lp[3];
  int tid = threadIdx.x;
  for (int i = tid; i < N; i += 256) d2[i] = 1e10f;
  if (tid == 0) { out[b * S + 0] = 0; lp[0] = P[0]; lp[1] = P[1]; lp[2] = P[2]; }
  __syncthreads();
  for (int s = 1; s < S; ++s) {
    float bestv = -1.f; int besti = 0x7fffffff;
    for (int i = tid; i < N; i += 256) {
      float dx = P[i * 3 + 0] - lp[0];
      float dy = P[i * 3 + 1] - lp[1];
      float dz = P[i * 3 + 2] - lp[2];
      float nd = fminf(d2[i], dx * dx + dy * dy + dz * dz);
      d2[i] = nd;
      if (nd > bestv) { bestv = nd; besti = i; }
    }
    rv[tid] = bestv; ri[tid] = besti;
    __syncthreads();
    for (int off = 128; off > 0; off >>= 1) {
      if (tid < off) {
        float v2 = rv[tid + off]; int i2 = ri[tid + off];
        if (v2 > rv[tid] || (v2 == rv[tid] && i2 < ri[tid])) { rv[tid] = v2; ri[tid] = i2; }
      }
      __syncthreads();
    }
    if (tid == 0) {
      int bi = ri[0];
      out[b * S + s] = bi;
      lp[0] = P[bi * 3 + 0]; lp[1] = P[bi * 3 + 1]; lp[2] = P[bi * 3 + 2];
    }
    __syncthreads();
  }
}

// new_xyz[b,s,:] = xyz[b, fidx[b,s], :]
__global__ void gather_xyz_kernel(const float* __restrict__ xyz,
                                  const int* __restrict__ fidx,
                                  float* __restrict__ nxyz, int Nin, int S)
{
  int i = blockIdx.x * blockDim.x + threadIdx.x;   // b*S+s
  if (i >= 16 * S) return;
  int b = i / S;
  int src = fidx[i];
  const float* p = xyz + ((size_t)b * Nin + src) * 3;
  nxyz[(size_t)i * 3 + 0] = p[0];
  nxyz[(size_t)i * 3 + 1] = p[1];
  nxyz[(size_t)i * 3 + 2] = p[2];
}

// KNN(K=32): one block per (b,s); 32 rounds of argmin (ties -> lowest idx)
__global__ __launch_bounds__(256) void knn_kernel(const float* __restrict__ new_xyz,
                                                  const float* __restrict__ xyz,
                                                  int Nin, int S, int* __restrict__ out)
{
  int bs = blockIdx.x;
  int b = bs / S;
  __shared__ float d[1024];
  __shared__ float rv[256];
  __shared__ int   ri[256];
  int tid = threadIdx.x;
  float qx = new_xyz[(size_t)bs * 3 + 0];
  float qy = new_xyz[(size_t)bs * 3 + 1];
  float qz = new_xyz[(size_t)bs * 3 + 2];
  const float* P = xyz + (size_t)b * Nin * 3;
  for (int i = tid; i < Nin; i += 256) {
    float dx = P[i * 3 + 0] - qx, dy = P[i * 3 + 1] - qy, dz = P[i * 3 + 2] - qz;
    d[i] = dx * dx + dy * dy + dz * dz;
  }
  __syncthreads();
  for (int k = 0; k < 32; ++k) {
    float bestv = 3.0e38f; int besti = 0x7fffffff;
    for (int i = tid; i < Nin; i += 256) {
      float v = d[i];
      if (v < bestv) { bestv = v; besti = i; }
    }
    rv[tid] = bestv; ri[tid] = besti;
    __syncthreads();
    for (int off = 128; off > 0; off >>= 1) {
      if (tid < off) {
        float v2 = rv[tid + off]; int i2 = ri[tid + off];
        if (v2 < rv[tid] || (v2 == rv[tid] && i2 < ri[tid])) { rv[tid] = v2; ri[tid] = i2; }
      }
      __syncthreads();
    }
    if (tid == 0) { out[(size_t)bs * 32 + k] = ri[0]; d[ri[0]] = 3.0e38f; }
    __syncthreads();
  }
}

// Gather neighbors, center over K, store cen[B,S,K,lc+3] (f32),
// accumulate per-batch sum/sumsq for ddof=1 std.
__global__ __launch_bounds__(128) void group_stats_kernel(
    const bf16_t* __restrict__ feat, const float* __restrict__ xyz,
    const int* __restrict__ nidx, float* __restrict__ cen,
    float* __restrict__ stats, int Nin, int S, int lc)
{
  const int C = lc + 3;
  const int bs = blockIdx.x;
  const int b = bs / S;
  __shared__ int sidx[32];
  if (threadIdx.x < 32) sidx[threadIdx.x] = nidx[(size_t)bs * 32 + threadIdx.x];
  __syncthreads();
  float lsum = 0.f, lsq = 0.f;
  for (int c = threadIdx.x; c < C; c += 128) {
    float m = 0.f;
    for (int k = 0; k < 32; ++k) {
      int idx = sidx[k];
      float v = (c < lc) ? (float)feat[((size_t)b * Nin + idx) * lc + c]
                         : xyz[((size_t)b * Nin + idx) * 3 + (c - lc)];
      m += v;
    }
    m *= (1.f / 32.f);
    for (int k = 0; k < 32; ++k) {
      int idx = sidx[k];
      float v = (c < lc) ? (float)feat[((size_t)b * Nin + idx) * lc + c]
                         : xyz[((size_t)b * Nin + idx) * 3 + (c - lc)];
      float cv = v - m;
      cen[((size_t)bs * 32 + k) * C + c] = cv;
      lsum += cv; lsq += cv * cv;
    }
  }
  __shared__ float s1[128], s2[128];
  s1[threadIdx.x] = lsum; s2[threadIdx.x] = lsq;
  __syncthreads();
  for (int off = 64; off > 0; off >>= 1) {
    if (threadIdx.x < off) {
      s1[threadIdx.x] += s1[threadIdx.x + off];
      s2[threadIdx.x] += s2[threadIdx.x + off];
    }
    __syncthreads();
  }
  if (threadIdx.x == 0) {
    atomicAdd(&stats[b * 2 + 0], s1[0]);
    atomicAdd(&stats[b * 2 + 1], s2[0]);
  }
}

__global__ void batch_std_kernel(const float* __restrict__ stats, float n,
                                 float* __restrict__ stdv)
{
  int b = threadIdx.x;
  if (b >= 16) return;
  float s = stats[b * 2 + 0], sq = stats[b * 2 + 1];
  float var = (sq - s * s / n) / (n - 1.f);
  stdv[b] = sqrtf(fmaxf(var, 0.f));
}

// h[row, 0:lc+3]       = alpha*(cen/(std+1e-5))+beta
// h[row, lc+3:2lc+3]   = feat[b, fidx[b,s], :] broadcast over k
// h[row, 2lc+3:CinPad] = 0   (zero padding for the 64-aligned GEMM K dim)
__global__ void build_group_input_kernel(
    const float* __restrict__ cen, const bf16_t* __restrict__ feat,
    const int* __restrict__ fidx, const float* __restrict__ stdv,
    const float* __restrict__ alpha, const float* __restrict__ beta,
    bf16_t* __restrict__ h, int Nin, int S, int lc, int CinPad)
{
  size_t i = (size_t)blockIdx.x * blockDim.x + threadIdx.x;
  const int C = lc + 3;
  size_t total = (size_t)16 * S * 32 * CinPad;
  if (i >= total) return;
  int c = (int)(i % CinPad);
  size_t row = i / CinPad;     // ((b*S)+s)*32 + k
  int bs = (int)(row >> 5);
  int b = bs / S;
  float y;
  if (c < C) {
    y = alpha[c] * (cen[row * C + c] / (stdv[b] + 1e-5f)) + beta[c];
  } else if (c < 2 * lc + 3) {
    int fi = fidx[bs];
    y = (float)feat[((size_t)b * Nin + fi) * lc + (c - C)];
  } else {
    y = 0.f;
  }
  h[i] = (bf16_t)y;
}

// max over K=32 rows: in [rows*32, C] -> out [rows, C]
__global__ void maxpool_k_kernel(const bf16_t* __restrict__ in,
                                 bf16_t* __restrict__ out, int rows, int C)
{
  size_t i = (size_t)blockIdx.x * blockDim.x + threadIdx.x;
  if (i >= (size_t)rows * C) return;
  int c = (int)(i % C);
  size_t r = i / C;
  const bf16_t* p = in + (r * 32) * (size_t)C + c;
  float m = -3.4e38f;
  for (int k = 0; k < 32; ++k) m = fmaxf(m, (float)p[(size_t)k * C]);
  out[i] = (bf16_t)m;
}

// global max over S: in [16, Srows, C] -> out [16, C]
__global__ void maxpool_s_kernel(const bf16_t* __restrict__ in,
                                 bf16_t* __restrict__ out, int Srows, int C)
{
  int i = blockIdx.x * blockDim.x + threadIdx.x;
  if (i >= 16 * C) return;
  int c = i % C, b = i / C;
  const bf16_t* p = in + ((size_t)b * Srows) * C + c;
  float m = -3.4e38f;
  for (int s = 0; s < Srows; ++s) m = fmaxf(m, (float)p[(size_t)s * C]);
  out[i] = (bf16_t)m;
}

// f32 [O,I] -> bf16 [Opad,Ipad], zero padded
__global__ void convert_weight_kernel(const float* __restrict__ src,
                                      bf16_t* __restrict__ dst,
                                      int O, int I, int Opad, int Ipad)
{
  size_t i = (size_t)blockIdx.x * blockDim.x + threadIdx.x;
  if (i >= (size_t)Opad * Ipad) return;
  int ii = (int)(i % Ipad);
  int o = (int)(i / Ipad);
  float v = (o < O && ii < I) ? src[(size_t)o * I + ii] : 0.f;
  dst[i] = (bf16_t)v;
}

__global__ void zero_kernel(float* p, int n)
{
  int i = blockIdx.x * blockDim.x + threadIdx.x;
  if (i < n) p[i] = 0.f;
}

// ===========================================================================
// Host orchestration
// ===========================================================================
// Input flattening (setup_inputs insertion order):
//  0: x [16,3,1024]
//  1..4: embed W,b,g,be
//  stage s (s=0..3), base = 5 + 38*s:
//    +0 alpha, +1 beta,
//    transfer: +2 W, +3 b, +4 g, +5 be
//    pre[0]:  +6 W1, +7 b1, +8 g1, +9 be1, +10 W2, +11 b2, +12 g2, +13 be2
//    pre[1]:  +14..+21   pos[0]: +22..+29   pos[1]: +30..+37
//  cls base 157: l1 W,b,g,be = 157..160; l2 = 161..164; l3 W=165, b=166
extern "C" void kernel_launch(void* const* d_in, const int* in_sizes, int n_in,
                              void* d_out, int out_size, void* d_ws, size_t ws_size,
                              hipStream_t stream)
{
  (void)in_sizes; (void)n_in; (void)out_size; (void)ws_size;
  const int Ns[5]  = {1024, 512, 256, 128, 64};
  const int lcs[4] = {64, 128, 256, 512};

  char* wsb = (char*)d_ws;
  size_t off = 0;
  auto alloc = [&](size_t bytes) -> void* {
    off = (off + 255) & ~(size_t)255;
    void* p = wsb + off;
    off += bytes;
    return p;
  };
  auto f32 = [&](int i) { return (const float*)d_in[i]; };

  // ---- workspace layout ----
  float*  xyzA  = (float*)alloc((size_t)16 * 1024 * 3 * 4);
  float*  xyzB  = (float*)alloc((size_t)16 * 1024 * 3 * 4);
  bf16_t* featA = (bf16_t*)alloc((size_t)16 * 1024 * 64 * 2);  // 1M elems all stages
  bf16_t* featB = (bf16_t*)alloc((size_t)16 * 1024 * 64 * 2);
  int*    fidx  = (int*)alloc((size_t)16 * 512 * 4);
  int*    nidx  = (int*)alloc((size_t)16 * 512 * 32 * 4);
  float*  stats = (float*)alloc(32 * 4);
  float*  stdv  = (float*)alloc(16 * 4);
  float*  cen   = (float*)alloc((size_t)16 * 512 * 32 * 67 * 4);   // worst: stage 1
  bf16_t* hbuf  = (bf16_t*)alloc((size_t)16 * 512 * 32 * 192 * 2); // Cin 64-aligned
  bf16_t* actT  = (bf16_t*)alloc((size_t)16 * 512 * 32 * 128 * 2);
  bf16_t* actM  = (bf16_t*)alloc((size_t)16 * 512 * 32 * 128 * 2);
  bf16_t* actR  = (bf16_t*)alloc((size_t)16 * 512 * 32 * 128 * 2);
  bf16_t* pool0 = (bf16_t*)alloc((size_t)16 * 512 * 128 * 2);
  bf16_t* pool1 = (bf16_t*)alloc((size_t)16 * 512 * 128 * 2);
  bf16_t* pool2 = (bf16_t*)alloc((size_t)16 * 512 * 128 * 2);
  bf16_t* gfeat = (bf16_t*)alloc((size_t)16 * 1024 * 2);
  bf16_t* c1    = (bf16_t*)alloc((size_t)16 * 512 * 2);
  bf16_t* c2    = (bf16_t*)alloc((size_t)16 * 256 * 2);

  auto convW = [&](int idx, int O, int I, int Opad, int Ipad) -> bf16_t* {
    bf16_t* dst = (bf16_t*)alloc((size_t)Opad * Ipad * 2);
    size_t tot = (size_t)Opad * Ipad;
    convert_weight_kernel<<<(unsigned)((tot + 255) / 256), 256, 0, stream>>>(
        f32(idx), dst, O, I, Opad, Ipad);
    return dst;
  };

  auto gemm = [&](const bf16_t* X, const bf16_t* W, const float* bias,
                  const float* g, const float* be, const bf16_t* Res,
                  float* Y32, bf16_t* Ybf, int M, int Cin, int Cout, int mode) {
    int coutPad = (Cout + 63) & ~63;
    dim3 grid(coutPad / 64, (M + 127) / 128);
    gemm_bf16_wmma<<<grid, 256, 0, stream>>>(X, W, bias, g, be, Res, Y32, Ybf,
                                             M, Cin, Cout, mode);
  };

  // ---- convert all weights to bf16 (zero-padded to 64-aligned shapes) ----
  bf16_t* WT[4];
  bf16_t* WR[4][8];
  for (int s = 0; s < 4; ++s) {
    int lc = lcs[s], oc = 2 * lc;
    int cinp = (2 * lc + 3 + 63) & ~63;
    int bI = 5 + s * 38;
    WT[s] = convW(bI + 2, oc, 2 * lc + 3, oc, cinp);
    const int wi[8] = {bI + 6, bI + 10, bI + 14, bI + 18, bI + 22, bI + 26, bI + 30, bI + 34};
    for (int j = 0; j < 8; ++j) WR[s][j] = convW(wi[j], oc, oc, oc, oc);
  }
  bf16_t* WL1 = convW(157, 512, 1024, 512, 1024);
  bf16_t* WL2 = convW(161, 256, 512, 256, 512);
  bf16_t* WL3 = convW(165, 40, 256, 64, 256);

  // ---- embedding ----
  transpose_xyz_kernel<<<(16 * 1024 + 255) / 256, 256, 0, stream>>>(f32(0), xyzA, 1024);
  embed_kernel<<<(16 * 1024 * 64 + 255) / 256, 256, 0, stream>>>(
      xyzA, f32(1), f32(2), f32(3), f32(4), featA, 1024);

  float*  xyzCur = xyzA;  float*  xyzNew = xyzB;
  bf16_t* featCur = featA; bf16_t* featNext = featB;

  // ---- stages ----
  for (int s = 0; s < 4; ++s) {
    const int lc = lcs[s], oc = 2 * lc, Nin = Ns[s], S = Ns[s + 1];
    const int C = lc + 3;
    const int cinp = (2 * lc + 3 + 63) & ~63;
    const int bI = 5 + s * 38;

    zero_kernel<<<1, 32, 0, stream>>>(stats, 32);
    fps_kernel<<<16, 256, 0, stream>>>(xyzCur, Nin, S, fidx);
    gather_xyz_kernel<<<(16 * S + 255) / 256, 256, 0, stream>>>(xyzCur, fidx, xyzNew, Nin, S);
    knn_kernel<<<16 * S, 256, 0, stream>>>(xyzNew, xyzCur, Nin, S, nidx);
    group_stats_kernel<<<16 * S, 128, 0, stream>>>(featCur, xyzCur, nidx, cen, stats, Nin, S, lc);
    batch_std_kernel<<<1, 16, 0, stream>>>(stats, (float)((size_t)S * 32 * C), stdv);

    size_t hTot = (size_t)16 * S * 32 * cinp;
    build_group_input_kernel<<<(unsigned)((hTot + 255) / 256), 256, 0, stream>>>(
        cen, featCur, fidx, stdv, f32(bI + 0), f32(bI + 1), hbuf, Nin, S, lc, cinp);

    const int M = 16 * S * 32;
    // transfer: CBR
    gemm(hbuf, WT[s], f32(bI + 3), f32(bI + 4), f32(bI + 5), nullptr, nullptr, actT, M, cinp, oc, 0);
    // pre[0]
    gemm(actT, WR[s][0], f32(bI + 7),  f32(bI + 8),  f32(bI + 9),  nullptr, nullptr, actM, M, oc, oc, 0);
    gemm(actM, WR[s][1], f32(bI + 11), f32(bI + 12), f32(bI + 13), actT,    nullptr, actR, M, oc, oc, 1);
    // pre[1]
    gemm(actR, WR[s][2], f32(bI + 15), f32(bI + 16), f32(bI + 17), nullptr, nullptr, actM, M, oc, oc, 0);
    gemm(actM, WR[s][3], f32(bI + 19), f32(bI + 20), f32(bI + 21), actR,    nullptr, actT, M, oc, oc, 1);
    // max over K
    const int rows = 16 * S;
    maxpool_k_kernel<<<(unsigned)(((size_t)rows * oc + 255) / 256), 256, 0, stream>>>(
        actT, pool0, rows, oc);
    // pos[0]
    gemm(pool0, WR[s][4], f32(bI + 23), f32(bI + 24), f32(bI + 25), nullptr, nullptr, pool1, rows, oc, oc, 0);
    gemm(pool1, WR[s][5], f32(bI + 27), f32(bI + 28), f32(bI + 29), pool0,  nullptr, pool2, rows, oc, oc, 1);
    // pos[1]
    gemm(pool2, WR[s][6], f32(bI + 31), f32(bI + 32), f32(bI + 33), nullptr, nullptr, pool1, rows, oc, oc, 0);
    gemm(pool1, WR[s][7], f32(bI + 35), f32(bI + 36), f32(bI + 37), pool2,  nullptr, featNext, rows, oc, oc, 1);

    { float* t = xyzCur; xyzCur = xyzNew; xyzNew = t; }
    { bf16_t* t = featCur; featCur = featNext; featNext = t; }
  }

  // ---- classifier ----
  maxpool_s_kernel<<<(16 * 1024 + 255) / 256, 256, 0, stream>>>(featCur, gfeat, 64, 1024);
  gemm(gfeat, WL1, f32(158), f32(159), f32(160), nullptr, nullptr, c1, 16, 1024, 512, 0);
  gemm(c1,    WL2, f32(162), f32(163), f32(164), nullptr, nullptr, c2, 16, 512, 256, 0);
  gemm(c2,    WL3, f32(166), f32(166), f32(166), nullptr, (float*)d_out, nullptr, 16, 256, 40, 2);
}
